// IDIR_69209103007764
// MI455X (gfx1250) — compile-verified, matrix operand order checked by reading, not run on hardware
//
#include <hip/hip_runtime.h>
#include <hip/hip_bf16.h>
#include <math.h>

// Problem constants (match reference)
#define TOK   1024      // B*S
#define HD    512       // H
#define ED    4         // experts
#define FD    2048      // expert hidden
#define RHD   2560      // reasoning hidden
#define VD    50000     // vocab
#define KSEL  32        // top-k in reasoning
#define MAXIT 20
#define TOLV  1e-3f

// ---------------- types ----------------
typedef __attribute__((ext_vector_type(16))) __bf16        v16bf;
typedef __attribute__((ext_vector_type(8)))  float         v8f;
typedef __attribute__((ext_vector_type(4)))  unsigned int  u32x4;

union Frag16 { v16bf v; u32x4 q[2]; };

__device__ __forceinline__ unsigned short f2bf(float f) {
  unsigned int u = __float_as_uint(f);
  u += 0x7FFFu + ((u >> 16) & 1u);        // round-to-nearest-even
  return (unsigned short)(u >> 16);
}

// gfx1250 async global->LDS copy (128b per lane), tracked by ASYNCcnt
__device__ __forceinline__ void async_g2l_b128(unsigned lds_off, const void* gaddr) {
  asm volatile("global_load_async_to_lds_b128 %0, %1, off"
               :: "v"(lds_off), "v"(gaddr) : "memory");
}
__device__ __forceinline__ void wait_async0() {
  asm volatile("s_wait_asynccnt 0x0" ::: "memory");
}

// ---------------- bf16 WMMA GEMM ----------------
// C[MxN] = op(A[MxK] @ Bt[NxK]^T (+bias[n]) (+addend[MxN])), op = relu?
// A row-major (M,K) bf16; Bt row-major (N,K) bf16 (i.e. torch-style weight).
// Requires: M % 128 == 0, K % 32 == 0. N arbitrary (guarded).
#define BM 128
#define BN 64
#define KC 32
#define STH 40   // padded LDS row stride in halves (KC + 8)
#define STU4 5   // same, in uint4 units

__global__ void gemm_bf16_wmma(const unsigned short* __restrict__ A,
                               const unsigned short* __restrict__ Bt,
                               float* __restrict__ Cf,
                               unsigned short* __restrict__ Cbf,
                               const float* __restrict__ bias,
                               const float* __restrict__ addend,
                               int M, int N, int K, int relu) {
  __shared__ u32x4 ldsA4[BM * STU4];   // 128 rows x 32 halves (padded) = 10 KB
  __shared__ u32x4 ldsB4[BN * STU4];   //  64 rows x 32 halves (padded) =  5 KB
  unsigned short* ldsA = (unsigned short*)ldsA4;
  unsigned short* ldsB = (unsigned short*)ldsB4;

  const int tid  = threadIdx.x;
  const int lane = tid & 31;
  const int wave = tid >> 5;     // 0..7
  const int wm   = wave & 3;     // 4 row groups of 32
  const int wn   = wave >> 2;    // 2 col groups of 32
  const int half = lane >> 4;
  const int lm   = lane & 15;

  const int blockM = blockIdx.y * BM;
  const int blockN = blockIdx.x * BN;

  // per-thread cooperative-load coordinates (one b128 per tile-chunk each)
  const int r  = tid >> 2;          // 0..63
  const int q  = tid & 3;           // uint4 within 32-half row
  const unsigned short* gA0 = A + (size_t)(blockM + r) * K + q * 8;
  const unsigned short* gA1 = gA0 + (size_t)64 * K;
  const int gn = blockN + r;
  const unsigned short* gB = Bt + (size_t)gn * K + q * 8;
  const unsigned ldsA0 = (unsigned)(uintptr_t)&ldsA4[r * STU4 + q];
  const unsigned ldsA1 = (unsigned)(uintptr_t)&ldsA4[(r + 64) * STU4 + q];
  const unsigned ldsB0 = (unsigned)(uintptr_t)&ldsB4[r * STU4 + q];

  v8f acc00 = {}, acc01 = {}, acc10 = {}, acc11 = {};

  const int ar0 = (wm * 32 + lm) * STH + half * 8;
  const int ar1 = ar0 + 16 * STH;
  const int br0 = (wn * 32 + lm) * STH + half * 8;
  const int br1 = br0 + 16 * STH;

  for (int k0 = 0; k0 < K; k0 += KC) {
    async_g2l_b128(ldsA0, gA0 + k0);
    async_g2l_b128(ldsA1, gA1 + k0);
    if (gn < N) {
      async_g2l_b128(ldsB0, gB + k0);
    } else {
      u32x4 zz = {0u, 0u, 0u, 0u};
      ldsB4[r * STU4 + q] = zz;      // zero-fill guard rows (rare partial block)
    }
    wait_async0();
    __syncthreads();

    Frag16 a0, a1, b0, b1;
    a0.q[0] = *(const u32x4*)(ldsA + ar0);
    a0.q[1] = *(const u32x4*)(ldsA + ar0 + 16);
    a1.q[0] = *(const u32x4*)(ldsA + ar1);
    a1.q[1] = *(const u32x4*)(ldsA + ar1 + 16);
    b0.q[0] = *(const u32x4*)(ldsB + br0);
    b0.q[1] = *(const u32x4*)(ldsB + br0 + 16);
    b1.q[0] = *(const u32x4*)(ldsB + br1);
    b1.q[1] = *(const u32x4*)(ldsB + br1 + 16);

    acc00 = __builtin_amdgcn_wmma_f32_16x16x32_bf16(false, a0.v, false, b0.v, (short)0, acc00, false, false);
    acc01 = __builtin_amdgcn_wmma_f32_16x16x32_bf16(false, a0.v, false, b1.v, (short)0, acc01, false, false);
    acc10 = __builtin_amdgcn_wmma_f32_16x16x32_bf16(false, a1.v, false, b0.v, (short)0, acc10, false, false);
    acc11 = __builtin_amdgcn_wmma_f32_16x16x32_bf16(false, a1.v, false, b1.v, (short)0, acc11, false, false);
    __syncthreads();
  }

  // epilogue: C/D fragment -> (row = vgpr + half*8, col = lane%16)
  const int col0 = blockN + wn * 32 + lm;
  const int col1 = col0 + 16;
  const float bv0 = (bias && col0 < N) ? bias[col0] : 0.f;
  const float bv1 = (bias && col1 < N) ? bias[col1] : 0.f;
#pragma unroll
  for (int rr = 0; rr < 8; ++rr) {
    int rowA = blockM + wm * 32 + half * 8 + rr;
    int rowB = rowA + 16;
    float v00 = acc00[rr] + bv0, v01 = acc01[rr] + bv1;
    float v10 = acc10[rr] + bv0, v11 = acc11[rr] + bv1;
    if (addend) {
      if (col0 < N) { v00 += addend[(size_t)rowA * N + col0]; v10 += addend[(size_t)rowB * N + col0]; }
      if (col1 < N) { v01 += addend[(size_t)rowA * N + col1]; v11 += addend[(size_t)rowB * N + col1]; }
    }
    if (relu) {
      v00 = fmaxf(v00, 0.f); v01 = fmaxf(v01, 0.f);
      v10 = fmaxf(v10, 0.f); v11 = fmaxf(v11, 0.f);
    }
    if (Cf) {
      if (col0 < N) { Cf[(size_t)rowA * N + col0] = v00; Cf[(size_t)rowB * N + col0] = v10; }
      if (col1 < N) { Cf[(size_t)rowA * N + col1] = v01; Cf[(size_t)rowB * N + col1] = v11; }
    }
    if (Cbf) {
      if (col0 < N) { Cbf[(size_t)rowA * N + col0] = f2bf(v00); Cbf[(size_t)rowB * N + col0] = f2bf(v10); }
      if (col1 < N) { Cbf[(size_t)rowA * N + col1] = f2bf(v01); Cbf[(size_t)rowB * N + col1] = f2bf(v11); }
    }
  }
}

// ---------------- Spectral norm (power iteration), 1 block x 512 ----------------
__device__ float blk_reduce_sum512(float* red, int t, float val) {
  red[t] = val;
  __syncthreads();
  for (int s = 256; s > 0; s >>= 1) {
    if (t < s) red[t] += red[t + s];
    __syncthreads();
  }
  float rv = red[0];
  __syncthreads();
  return rv;
}

__global__ void spectral_kernel(const float* __restrict__ W, float* __restrict__ sigma_out) {
  __shared__ float u[HD], v[HD], red[HD];
  const int t = threadIdx.x;
  u[t] = rsqrtf((float)HD);
  __syncthreads();
  for (int it = 0; it < 30; ++it) {
    float s = 0.f;
    for (int i = 0; i < HD; ++i) s += W[(size_t)i * HD + t] * u[i];   // v = W^T u
    float n = blk_reduce_sum512(red, t, s * s);
    v[t] = s / (sqrtf(n) + 1e-12f);
    __syncthreads();
    float s2 = 0.f;
    for (int j = 0; j < HD; ++j) s2 += W[(size_t)t * HD + j] * v[j];  // u = W v
    float n2 = blk_reduce_sum512(red, t, s2 * s2);
    u[t] = s2 / (sqrtf(n2) + 1e-12f);
    __syncthreads();
  }
  float s = 0.f;
  for (int i = 0; i < HD; ++i) s += W[(size_t)i * HD + t] * u[i];     // v = W^T u
  float n = blk_reduce_sum512(red, t, s * s);
  v[t] = s / (sqrtf(n) + 1e-12f);
  __syncthreads();
  float wv = 0.f;
  for (int j = 0; j < HD; ++j) wv += W[(size_t)t * HD + j] * v[j];    // (W v)[t]
  float sigma = blk_reduce_sum512(red, t, u[t] * wv);
  if (t == 0) *sigma_out = sigma;
}

// ---------------- Converters ----------------
// straight f32 -> bf16 (optional scale = 1/(*scale_ptr))
__global__ void conv_f32_bf16(const float* __restrict__ src, unsigned short* __restrict__ dst,
                              int n, const float* __restrict__ scale_ptr) {
  int i = blockIdx.x * 256 + threadIdx.x;
  if (i >= n) return;
  float s = scale_ptr ? (1.0f / *scale_ptr) : 1.0f;
  dst[i] = f2bf(src[i] * s);
}

// src (R x C) f32 row-major -> dst (C x R) bf16 row-major
__global__ void trans_f32_bf16(const float* __restrict__ src, unsigned short* __restrict__ dst,
                               int R, int C) {
  int i = blockIdx.x * 256 + threadIdx.x;
  if (i >= R * C) return;
  int r = i / C, c = i % C;
  dst[(size_t)c * R + r] = f2bf(src[i]);
}

__global__ void gather_embed(const int* __restrict__ x, const float* __restrict__ emb,
                             unsigned short* __restrict__ e_bf) {
  int i = blockIdx.x * 256 + threadIdx.x;  // TOK*HD
  int t = i >> 9, c = i & 511;
  e_bf[i] = f2bf(emb[(size_t)x[t] * HD + c]);
}

__global__ void init_h(float* h, unsigned short* hbf) {
  int i = blockIdx.x * 256 + threadIdx.x;
  h[i] = 0.f;
  hbf[i] = 0;
}

__global__ void init_state(float* norm, int* done, int* iters) {
  *norm = 0.f; *done = 0; *iters = 0;
}

// ---------------- Router: softmax over E=4, top-2, normalized coefficients ----------------
__global__ void router_coef(const float* __restrict__ h, const float* __restrict__ router_w,
                            float* __restrict__ coef) {
  int t = blockIdx.x * 256 + threadIdx.x;
  if (t >= TOK) return;
  float g[ED];
  for (int e = 0; e < ED; ++e) {
    float s = 0.f;
    for (int k = 0; k < HD; ++k) s += h[(size_t)t * HD + k] * router_w[(size_t)e * HD + k];
    g[e] = s;
  }
  float m = g[0];
  for (int e = 1; e < ED; ++e) m = fmaxf(m, g[e]);
  float ssum = 0.f;
  for (int e = 0; e < ED; ++e) { g[e] = __expf(g[e] - m); ssum += g[e]; }
  for (int e = 0; e < ED; ++e) g[e] /= ssum;
  int i0 = 0;
  for (int e = 1; e < ED; ++e) if (g[e] > g[i0]) i0 = e;          // ties -> lowest idx
  int i1 = -1;
  for (int e = 0; e < ED; ++e) if (e != i0 && (i1 < 0 || g[e] > g[i1])) i1 = e;
  float w0 = g[i0], w1 = g[i1], tsum = w0 + w1;
  for (int e = 0; e < ED; ++e)
    coef[(size_t)t * ED + e] = (e == i0) ? w0 / tsum : (e == i1) ? w1 / tsum : 0.f;
}

__global__ void moe_acc(float* __restrict__ moe, const float* __restrict__ eo2,
                        const float* __restrict__ coef, int e, int init) {
  int i = blockIdx.x * 256 + threadIdx.x;  // TOK*HD
  int t = i >> 9;
  float base = init ? 0.f : moe[i];
  moe[i] = base + coef[(size_t)t * ED + e] * eo2[i];
}

// zh = [z, h] as bf16 (TOK x 2H)
__global__ void concat_zh(const float* __restrict__ z, const float* __restrict__ h,
                          unsigned short* __restrict__ zh) {
  int i = blockIdx.x * 256 + threadIdx.x;  // TOK*2H
  int t = i >> 10, j = i & 1023;
  float v = (j < HD) ? z[(size_t)t * HD + j] : h[(size_t)t * HD + (j - HD)];
  zh[i] = f2bf(v);
}

// ---------------- top-K(32) soft mask per token; z *= mask; also write z bf16 ----------------
__global__ void topk_mask(const float* __restrict__ scores, float* __restrict__ z,
                          unsigned short* __restrict__ zbf) {
  __shared__ float sc[HD];
  __shared__ float mask[HD];
  __shared__ float redv[256];
  __shared__ int   redi[256];
  __shared__ float topv[KSEL];
  __shared__ int   topi[KSEL];
  const int t = blockIdx.x, tid = threadIdx.x;
  sc[tid]       = scores[(size_t)t * HD + tid];
  sc[tid + 256] = scores[(size_t)t * HD + tid + 256];
  mask[tid] = 0.f; mask[tid + 256] = 0.f;
  __syncthreads();
  for (int it = 0; it < KSEL; ++it) {
    float v0 = sc[tid], v1 = sc[tid + 256];
    float bv; int bi;
    if (v1 > v0) { bv = v1; bi = tid + 256; } else { bv = v0; bi = tid; }
    redv[tid] = bv; redi[tid] = bi;
    __syncthreads();
    for (int s = 128; s > 0; s >>= 1) {
      if (tid < s) {
        if (redv[tid + s] > redv[tid] ||
            (redv[tid + s] == redv[tid] && redi[tid + s] < redi[tid])) {
          redv[tid] = redv[tid + s]; redi[tid] = redi[tid + s];
        }
      }
      __syncthreads();
    }
    if (tid == 0) { topv[it] = redv[0]; topi[it] = redi[0]; sc[redi[0]] = -3.4e38f; }
    __syncthreads();
  }
  if (tid == 0) {            // softmax over 32 (topv[0] is max) + scatter
    float ssum = 0.f, w[KSEL];
    for (int k = 0; k < KSEL; ++k) { w[k] = __expf(topv[k] - topv[0]); ssum += w[k]; }
    for (int k = 0; k < KSEL; ++k) mask[topi[k]] = w[k] / ssum;
  }
  __syncthreads();
  for (int j = tid; j < HD; j += 256) {
    float nz = z[(size_t)t * HD + j] * mask[j];
    z[(size_t)t * HD + j] = nz;
    zbf[(size_t)t * HD + j] = f2bf(nz);
  }
}

// ---------------- combine: h_new = relu(hW+ex+moe+reason); accumulate ||dh||^2 ----------------
__global__ void combine(const float* __restrict__ hW, const float* __restrict__ ex,
                        const float* __restrict__ moe, const float* __restrict__ reason,
                        const float* __restrict__ h_old, float* __restrict__ h_new,
                        unsigned short* __restrict__ h_new_bf,
                        float* __restrict__ norm_acc, const int* __restrict__ done,
                        int ignore_done) {
  __shared__ float red[256];
  int i = blockIdx.x * 256 + threadIdx.x;  // TOK*HD
  int frozen = (!ignore_done) && (*done != 0);
  float hv = frozen ? h_old[i] : fmaxf(hW[i] + ex[i] + moe[i] + reason[i], 0.f);
  h_new[i] = hv;
  h_new_bf[i] = f2bf(hv);
  float d = hv - h_old[i];
  red[threadIdx.x] = d * d;
  __syncthreads();
  for (int s = 128; s > 0; s >>= 1) {
    if (threadIdx.x < s) red[threadIdx.x] += red[threadIdx.x + s];
    __syncthreads();
  }
  if (threadIdx.x == 0) atomicAdd(norm_acc, red[0]);
}

__global__ void check_conv(float* norm_acc, int* done, int* iters, int k) {
  if (*done == 0) {
    if (sqrtf(*norm_acc) < TOLV) *done = 1;
    else *iters = k + 1;
  }
  *norm_acc = 0.f;
}

__global__ void write_iters(const int* iters, float* out, size_t pos) {
  ((int*)out)[pos] = *iters;
}

// ---------------- host orchestration ----------------
extern "C" void kernel_launch(void* const* d_in, const int* in_sizes, int n_in,
                              void* d_out, int out_size, void* d_ws, size_t ws_size,
                              hipStream_t stream) {
  const int*   x        = (const int*)  d_in[0];
  const float* emb      = (const float*)d_in[1];
  const float* W_h      = (const float*)d_in[2];
  const float* W_x      = (const float*)d_in[3];
  const float* router_w = (const float*)d_in[4];
  const float* exp_w1   = (const float*)d_in[5];
  const float* exp_w2   = (const float*)d_in[6];
  const float* T1_w     = (const float*)d_in[7];
  const float* T1_b     = (const float*)d_in[8];
  const float* T2_w     = (const float*)d_in[9];
  const float* T2_b     = (const float*)d_in[10];
  const float* R1_w     = (const float*)d_in[11];
  const float* R1_b     = (const float*)d_in[12];
  const float* R2_w     = (const float*)d_in[13];
  const float* R2_b     = (const float*)d_in[14];
  const float* out_w    = (const float*)d_in[15];
  const float* out_b    = (const float*)d_in[16];
  float* logits = (float*)d_out;

  // bump allocator over d_ws (~125 MB total)
  char* wp = (char*)d_ws;
  auto alloc = [&](size_t bytes) -> void* {
    void* p = (void*)wp;
    wp += (bytes + 255) & ~(size_t)255;
    return p;
  };
  float* sigma   = (float*)alloc(4);
  float* normacc = (float*)alloc(4);
  int*   done    = (int*)  alloc(4);
  int*   iters   = (int*)  alloc(4);
  // bf16 weights, all stored as (N x K) row-major ("torch layout")
  unsigned short* WhS  = (unsigned short*)alloc((size_t)HD * HD * 2);        // W_h / sigma
  unsigned short* Wx   = (unsigned short*)alloc((size_t)HD * HD * 2);
  unsigned short* ew1t = (unsigned short*)alloc((size_t)ED * FD * HD * 2);   // (F x H) per expert
  unsigned short* ew2t = (unsigned short*)alloc((size_t)ED * HD * FD * 2);   // (H x F) per expert
  unsigned short* T1   = (unsigned short*)alloc((size_t)RHD * 2 * HD * 2);   // (RH x 2H)
  unsigned short* T2   = (unsigned short*)alloc((size_t)HD * RHD * 2);       // (H x RH)
  unsigned short* R1   = (unsigned short*)alloc((size_t)RHD * HD * 2);       // (RH x H)
  unsigned short* R2   = (unsigned short*)alloc((size_t)HD * RHD * 2);       // (H x RH)
  unsigned short* Ow   = (unsigned short*)alloc((size_t)VD * HD * 2);        // (V x H)
  unsigned short* e_bf = (unsigned short*)alloc((size_t)TOK * HD * 2);
  float* ex    = (float*)alloc((size_t)TOK * HD * 4);
  float* hbuf[2];  unsigned short* hbf[2];
  hbuf[0] = (float*)alloc((size_t)TOK * HD * 4);
  hbuf[1] = (float*)alloc((size_t)TOK * HD * 4);
  hbf[0]  = (unsigned short*)alloc((size_t)TOK * HD * 2);
  hbf[1]  = (unsigned short*)alloc((size_t)TOK * HD * 2);
  float* hW     = (float*)alloc((size_t)TOK * HD * 4);
  float* coef   = (float*)alloc((size_t)TOK * ED * 4);
  unsigned short* eo1bf = (unsigned short*)alloc((size_t)TOK * FD * 2);
  float* eo2    = (float*)alloc((size_t)TOK * HD * 4);
  float* moe    = (float*)alloc((size_t)TOK * HD * 4);
  float* z      = (float*)alloc((size_t)TOK * HD * 4);
  unsigned short* zbf  = (unsigned short*)alloc((size_t)TOK * HD * 2);
  unsigned short* zh   = (unsigned short*)alloc((size_t)TOK * 2 * HD * 2);
  unsigned short* s1bf = (unsigned short*)alloc((size_t)TOK * RHD * 2);
  float* scores = (float*)alloc((size_t)TOK * HD * 4);
  unsigned short* r1bf = (unsigned short*)alloc((size_t)TOK * RHD * 2);
  float* reason = (float*)alloc((size_t)TOK * HD * 4);
  float* hstar  = (float*)alloc((size_t)TOK * HD * 4);
  unsigned short* hsbf = (unsigned short*)alloc((size_t)TOK * HD * 2);

  auto gemm = [&](const unsigned short* A, const unsigned short* Bt, float* Cf,
                  unsigned short* Cbf, const float* bias, const float* add,
                  int M, int N, int K, int relu) {
    dim3 grid((N + BN - 1) / BN, M / BM);
    gemm_bf16_wmma<<<grid, 256, 0, stream>>>(A, Bt, Cf, Cbf, bias, add, M, N, K, relu);
  };
  auto blocks = [](size_t n) { return (unsigned)((n + 255) / 256); };

  // ---- setup ----
  spectral_kernel<<<1, HD, 0, stream>>>(W_h, sigma);
  conv_f32_bf16<<<blocks((size_t)HD * HD), 256, 0, stream>>>(W_h, WhS, HD * HD, sigma);
  conv_f32_bf16<<<blocks((size_t)HD * HD), 256, 0, stream>>>(W_x, Wx, HD * HD, nullptr);
  for (int e = 0; e < ED; ++e) {
    // exp_w1[e] is (H x F) -> need (F x H); exp_w2[e] is (F x H) -> need (H x F)
    trans_f32_bf16<<<blocks((size_t)HD * FD), 256, 0, stream>>>(
        exp_w1 + (size_t)e * HD * FD, ew1t + (size_t)e * FD * HD, HD, FD);
    trans_f32_bf16<<<blocks((size_t)FD * HD), 256, 0, stream>>>(
        exp_w2 + (size_t)e * FD * HD, ew2t + (size_t)e * HD * FD, FD, HD);
  }
  conv_f32_bf16<<<blocks((size_t)RHD * 2 * HD), 256, 0, stream>>>(T1_w, T1, RHD * 2 * HD, nullptr);
  conv_f32_bf16<<<blocks((size_t)HD * RHD), 256, 0, stream>>>(T2_w, T2, HD * RHD, nullptr);
  conv_f32_bf16<<<blocks((size_t)RHD * HD), 256, 0, stream>>>(R1_w, R1, RHD * HD, nullptr);
  conv_f32_bf16<<<blocks((size_t)HD * RHD), 256, 0, stream>>>(R2_w, R2, HD * RHD, nullptr);
  conv_f32_bf16<<<blocks((size_t)VD * HD), 256, 0, stream>>>(out_w, Ow, VD * HD, nullptr);
  gather_embed<<<blocks((size_t)TOK * HD), 256, 0, stream>>>(x, emb, e_bf);
  init_h<<<blocks((size_t)TOK * HD), 256, 0, stream>>>(hbuf[0], hbf[0]);
  init_state<<<1, 1, 0, stream>>>(normacc, done, iters);
  gemm(e_bf, Wx, ex, nullptr, nullptr, nullptr, TOK, HD, HD, 0);   // ex = e @ W_x^T

  // ---- one recurrence step ----
  auto step = [&](const float* hin_f, const unsigned short* hin_b,
                  float* hout_f, unsigned short* hout_b, int ignore_done, int kk) {
    gemm(hin_b, WhS, hW, nullptr, nullptr, nullptr, TOK, HD, HD, 0);
    router_coef<<<blocks(TOK), 256, 0, stream>>>(hin_f, router_w, coef);
    for (int e = 0; e < ED; ++e) {
      gemm(hin_b, ew1t + (size_t)e * FD * HD, nullptr, eo1bf, nullptr, nullptr, TOK, FD, HD, 1);
      gemm(eo1bf, ew2t + (size_t)e * HD * FD, eo2, nullptr, nullptr, nullptr, TOK, HD, FD, 0);
      moe_acc<<<blocks((size_t)TOK * HD), 256, 0, stream>>>(moe, eo2, coef, e, e == 0);
    }
    hipMemcpyAsync(z, hin_f, (size_t)TOK * HD * 4, hipMemcpyDeviceToDevice, stream);
    for (int s = 0; s < 3; ++s) {
      concat_zh<<<blocks((size_t)TOK * 2 * HD), 256, 0, stream>>>(z, hin_f, zh);
      gemm(zh, T1, nullptr, s1bf, T1_b, nullptr, TOK, RHD, 2 * HD, 1);
      gemm(s1bf, T2, scores, nullptr, T2_b, nullptr, TOK, HD, RHD, 0);
      topk_mask<<<TOK, 256, 0, stream>>>(scores, z, zbf);
    }
    gemm(zbf, R1, nullptr, r1bf, R1_b, nullptr, TOK, RHD, HD, 1);
    gemm(r1bf, R2, reason, nullptr, R2_b, nullptr, TOK, HD, RHD, 0);
    combine<<<blocks((size_t)TOK * HD), 256, 0, stream>>>(hW, ex, moe, reason, hin_f,
                                                          hout_f, hout_b, normacc, done,
                                                          ignore_done);
    if (!ignore_done) check_conv<<<1, 1, 0, stream>>>(normacc, done, iters, kk);
  };

  for (int k = 0; k < MAXIT; ++k)
    step(hbuf[k & 1], hbf[k & 1], hbuf[(k + 1) & 1], hbf[(k + 1) & 1], 0, k);
  // h_star = step(h) with no freeze
  step(hbuf[MAXIT & 1], hbf[MAXIT & 1], hstar, hsbf, 1, MAXIT);

  // logits = h_star @ out_w^T + out_b   -> d_out[0 : TOK*VD]
  gemm(hsbf, Ow, logits, nullptr, out_b, nullptr, TOK, VD, HD, 0);
  write_iters<<<1, 1, 0, stream>>>(iters, logits, (size_t)TOK * VD);
}
